// QuantTransformerEncoderLayer_44555990729117
// MI455X (gfx1250) — matrix operand order repaired and checked by reading, not compiled, not run on hardware
//
#include <hip/hip_runtime.h>
#include <hip/hip_bf16.h>
#include <math.h>

// ---------------- types ----------------
typedef __attribute__((ext_vector_type(16))) _Float16 v16h;
typedef __attribute__((ext_vector_type(8)))  _Float16 v8h;
typedef __attribute__((ext_vector_type(8)))  float    v8f;

// ---------------- constants (problem shape) ----------------
#define BB 4
#define NN 512
#define DD 1024
#define HH 16
#define FFD 4096
#define DHD 64
#define MROWS (BB*NN)   // 2048

// ---------------- float atomic min/max (order-independent => deterministic) ----
__device__ inline void atomicMaxF(float* a, float v) {
    unsigned int* u = (unsigned int*)a;
    unsigned int old = *u;
    while (__uint_as_float(old) < v) {
        unsigned int assumed = old;
        old = atomicCAS(u, assumed, __float_as_uint(v));
        if (old == assumed) break;
    }
}
__device__ inline void atomicMinF(float* a, float v) {
    unsigned int* u = (unsigned int*)a;
    unsigned int old = *u;
    while (__uint_as_float(old) > v) {
        unsigned int assumed = old;
        old = atomicCAS(u, assumed, __float_as_uint(v));
        if (old == assumed) break;
    }
}

// ---------------- block reductions (blockDim.x == 256) ----------------
__device__ inline float blockRedSum(float v) {
    __shared__ float sh[256];
    int t = threadIdx.x;
    sh[t] = v; __syncthreads();
    for (int s = 128; s > 0; s >>= 1) { if (t < s) sh[t] += sh[t + s]; __syncthreads(); }
    float r = sh[0]; __syncthreads();
    return r;
}
__device__ inline float blockRedMax(float v) {
    __shared__ float sh[256];
    int t = threadIdx.x;
    sh[t] = v; __syncthreads();
    for (int s = 128; s > 0; s >>= 1) { if (t < s) sh[t] = fmaxf(sh[t], sh[t + s]); __syncthreads(); }
    float r = sh[0]; __syncthreads();
    return r;
}
__device__ inline float blockRedMin(float v) {
    __shared__ float sh[256];
    int t = threadIdx.x;
    sh[t] = v; __syncthreads();
    for (int s = 128; s > 0; s >>= 1) { if (t < s) sh[t] = fminf(sh[t], sh[t + s]); __syncthreads(); }
    float r = sh[0]; __syncthreads();
    return r;
}

// ---------------- scalar init ----------------
// scal layout (floats): 0/1 src min/max, 2/3 W_in, 4/5 q, 6/7 k, 8/9 v, 11 attn-max
__global__ void init_scalars_k(float* s) {
    int t = threadIdx.x;
    if (t < 16) s[t] = (t & 1) ? -__builtin_inff() : __builtin_inff();
}

// ---------------- min/max over a (rows x cols) view with leading dim ld ------
__global__ void minmax_k(const float* __restrict__ x, int rows, int cols, int ld,
                         float* scal, int mini, int maxi) {
    float mn = __builtin_inff(), mx = -__builtin_inff();
    unsigned int total = (unsigned int)rows * (unsigned int)cols;
    unsigned int stride = gridDim.x * blockDim.x;
    if (ld == cols) {
        for (unsigned int i = blockIdx.x * blockDim.x + threadIdx.x; i < total; i += stride) {
            float v = x[i];
            mn = fminf(mn, v); mx = fmaxf(mx, v);
        }
    } else {
        for (unsigned int i = blockIdx.x * blockDim.x + threadIdx.x; i < total; i += stride) {
            unsigned int r = i / (unsigned int)cols;
            unsigned int c = i - r * (unsigned int)cols;
            float v = x[(size_t)r * ld + c];
            mn = fminf(mn, v); mx = fmaxf(mx, v);
        }
    }
    mn = blockRedMin(mn);
    mx = blockRedMax(mx);
    if (threadIdx.x == 0) { atomicMinF(&scal[mini], mn); atomicMaxF(&scal[maxi], mx); }
}

// ---------------- fake-quant scalar op -----------------------------------
// mode 0: plain f32->f16; mode 1: affine (delta,zp); mode 2: always_zero (delta only)
__device__ inline _Float16 qh(float x, float d, float zp, int mode) {
    if (mode == 1) {
        float q = fminf(fmaxf(rintf(x / d) + zp, 0.f), 255.f);
        x = (q - zp) * d;
    } else if (mode == 2) {
        float q = fminf(fmaxf(rintf(x / d), 0.f), 255.f);
        x = q * d;
    }
    return (_Float16)x;
}
__device__ inline void qparams(const float* scal, int qi, int mode, float& d, float& zp) {
    d = 1.f; zp = 0.f;
    if (mode == 1) {
        float mn = scal[qi], mx = scal[qi + 1];
        d = fmaxf((mx - mn) * (1.f / 255.f), 1e-8f);
        zp = rintf(-mn / d);
    } else if (mode == 2) {
        d = fmaxf(scal[qi] * (1.f / 255.f), 1e-8f);
    }
}

// ---------------- staging: f32 -> (fake-quant) -> f16 ------------------------
__global__ void cvt_q_k(const float* __restrict__ in, int rows, int cols, int ldin,
                        _Float16* __restrict__ outp, int ldout,
                        const float* __restrict__ scal, int qi, int mode) {
    float d, zp;
    qparams(scal, qi, mode, d, zp);
    unsigned int total = (unsigned int)rows * (unsigned int)cols;
    unsigned int stride = gridDim.x * blockDim.x;
    if (ldin == cols && ldout == cols) {
        for (unsigned int i = blockIdx.x * blockDim.x + threadIdx.x; i < total; i += stride)
            outp[i] = qh(in[i], d, zp, mode);
    } else {
        for (unsigned int i = blockIdx.x * blockDim.x + threadIdx.x; i < total; i += stride) {
            unsigned int r = i / (unsigned int)cols;
            unsigned int c = i - r * (unsigned int)cols;
            outp[(size_t)r * ldout + c] = qh(in[(size_t)r * ldin + c], d, zp, mode);
        }
    }
}

// ---------------- per-head transpose+quant of V: qkv f32 -> vt f16 ----------
// In : qkv[(b*NN+m)*3DD + 2*DD + h*DHD + d]   (m=0..NN-1, d=0..DHD-1)
// Out: vt[((b*HH+h)*DHD + d)*NN + m]          (so PV GEMM uses contiguous NxK B)
__global__ void vtrans_k(const float* __restrict__ qkv, _Float16* __restrict__ vt,
                         const float* __restrict__ scal, int qi) {
    __shared__ float tile[16][17];
    float d, zp;
    qparams(scal, qi, 1, d, zp);
    int z = blockIdx.z, b = z / HH, h = z % HH;
    int m0 = blockIdx.x * 16, d0 = blockIdx.y * 16;
    int tx = threadIdx.x, ty = threadIdx.y;
    tile[ty][tx] = qkv[(size_t)(b * NN + m0 + ty) * (3 * DD) + 2 * DD + h * DHD + d0 + tx];
    __syncthreads();
    vt[((size_t)z * DHD + d0 + ty) * NN + m0 + tx] = qh(tile[tx][ty], d, zp, 1);
}

// ---------------- WMMA GEMM (all-f16 operands, f32 accum) --------------------
// C[z] = alpha * A[z] @ B[z]^T (+bias)
// A: f16 MxK row-major (lda); B: f16 NxK row-major (ldb)
// per-batch offsets: (z/cz)*s0 + (z%cz)*s1  (element units)
// epi: 0 plain, 1 *rmask[row], 2 relu then *rmask[row]; outHalf: store f16
// K % 32 == 0. Block: 128 threads (4 waves), tile 128x128, 64x64 per wave.
__launch_bounds__(128)
__global__ void gemm16_k(const _Float16* __restrict__ A, int lda, long long sA0, long long sA1, int czA,
                         const _Float16* __restrict__ Bm, int ldb, long long sB0, long long sB1, int czB,
                         float* __restrict__ C, int ldc, long long sC0, long long sC1, int czC,
                         int M, int N, int K,
                         const float* __restrict__ bias,
                         const float* __restrict__ rmask,
                         int epi, float alpha, int outHalf) {
    int z = blockIdx.z;
    const _Float16* Ab = A  + (long long)(z / czA) * sA0 + (long long)(z % czA) * sA1;
    const _Float16* Bb = Bm + (long long)(z / czB) * sB0 + (long long)(z % czB) * sB1;
    float*           Cb = C + (long long)(z / czC) * sC0 + (long long)(z % czC) * sC1;

    int lane = threadIdx.x & 31;
    int wv   = threadIdx.x >> 5;         // 0..3
    int wm   = wv >> 1, wn = wv & 1;
    int l16  = lane & 15, hi = lane >> 4;
    int rowBase = blockIdx.y * 128 + wm * 64;
    int colBase = blockIdx.x * 128 + wn * 64;

    v8f acc[4][4];
#pragma unroll
    for (int i = 0; i < 4; i++)
#pragma unroll
        for (int j = 0; j < 4; j++)
#pragma unroll
            for (int r = 0; r < 8; r++) acc[i][j][r] = 0.f;

    for (int k0 = 0; k0 < K; k0 += 32) {
        v16h af[4], bf[4];
        // A frag: lanes 0-15 take K {0..15}, lanes 16-31 take K {16..31}
#pragma unroll
        for (int mi = 0; mi < 4; mi++) {
            int row = rowBase + mi * 16 + l16; if (row >= M) row = M - 1;
            const _Float16* p = Ab + (long long)row * lda + k0 + hi * 16;
            v8h c0 = *(const v8h*)(p);
            v8h c1 = *(const v8h*)(p + 8);
            v16h r;
#pragma unroll
            for (int i = 0; i < 8; i++) { r[i] = c0[i]; r[i + 8] = c1[i]; }
            af[mi] = r;
        }
        // B frag: lane%16 = N column, 16 contiguous K halfs per lane (32B aligned)
#pragma unroll
        for (int ni = 0; ni < 4; ni++) {
            int col = colBase + ni * 16 + l16; if (col >= N) col = N - 1;
            const _Float16* p = Bb + (long long)col * ldb + k0 + hi * 16;
            bf[ni] = *(const v16h*)(p);
        }
#pragma unroll
        for (int mi = 0; mi < 4; mi++)
#pragma unroll
            for (int ni = 0; ni < 4; ni++)
                acc[mi][ni] = __builtin_amdgcn_wmma_f32_16x16x32_f16(
                    false, af[mi], false, bf[ni], (short)0, acc[mi][ni], false, false);
    }

    // store: C VGPR r holds M = r (lanes 0-15) / r+8 (lanes 16-31), N = lane%16
    bool full = (rowBase + 64 <= M) && (colBase + 64 <= N);
    if (full) {
#pragma unroll
        for (int mi = 0; mi < 4; mi++)
#pragma unroll
            for (int ni = 0; ni < 4; ni++) {
                int col = colBase + ni * 16 + l16;
#pragma unroll
                for (int r = 0; r < 8; r++) {
                    int row = rowBase + mi * 16 + hi * 8 + r;
                    float v = alpha * acc[mi][ni][r];
                    if (bias) v += bias[col];
                    if (epi == 1)      v *= rmask[row];
                    else if (epi == 2) v = fmaxf(v, 0.f) * rmask[row];
                    if (outHalf) ((_Float16*)Cb)[(long long)row * ldc + col] = (_Float16)v;
                    else         Cb[(long long)row * ldc + col] = v;
                }
            }
    } else {
#pragma unroll
        for (int mi = 0; mi < 4; mi++)
#pragma unroll
            for (int ni = 0; ni < 4; ni++) {
                int col = colBase + ni * 16 + l16;
#pragma unroll
                for (int r = 0; r < 8; r++) {
                    int row = rowBase + mi * 16 + hi * 8 + r;
                    if (row < M && col < N) {
                        float v = alpha * acc[mi][ni][r];
                        if (bias) v += bias[col];
                        if (epi == 1)      v *= rmask[row];
                        else if (epi == 2) v = fmaxf(v, 0.f) * rmask[row];
                        if (outHalf) ((_Float16*)Cb)[(long long)row * ldc + col] = (_Float16)v;
                        else         Cb[(long long)row * ldc + col] = v;
                    }
                }
            }
    }
}

// ---------------- masked softmax over 512-wide rows, in-place, track prob max
// aw batch z uses mask batch (z % B): replicates torch's h-major mask tiling.
__global__ void softmax_k(float* __restrict__ aw, const float* __restrict__ mask,
                          float* __restrict__ scal, int amaxi) {
    int row = blockIdx.x;             // 0 .. 64*512-1
    int z = row / NN, n = row - z * NN;
    int mb = z % BB;                  // h-major tiling quirk
    const float* mrow = mask + (size_t)mb * NN;
    float mq = mrow[n];
    float* p = aw + (size_t)row * NN;
    int t = threadIdx.x;
    float v[2];
    float lmax = -__builtin_inff();
#pragma unroll
    for (int j = 0; j < 2; j++) {
        int m = t + j * 256;
        float keep = mq * mrow[m];
        float x = (keep == 0.f) ? -10000.f : p[m];
        v[j] = x; lmax = fmaxf(lmax, x);
    }
    float rmax = blockRedMax(lmax);
    float ls = 0.f;
#pragma unroll
    for (int j = 0; j < 2; j++) { v[j] = expf(v[j] - rmax); ls += v[j]; }
    float rsum = blockRedSum(ls);
    float inv = 1.f / rsum;
    float pmax = 0.f;
#pragma unroll
    for (int j = 0; j < 2; j++) {
        float pr = v[j] * inv;
        p[t + j * 256] = pr;
        pmax = fmaxf(pmax, pr);
    }
    pmax = blockRedMax(pmax);
    if (t == 0) atomicMaxF(&scal[amaxi], pmax);
}

// ---------------- residual + LayerNorm (D=1024, block 256, one block per row)
// resid = x + y * (mask[row] * (useMask0 ? mask[b*N] : 1))
// out = LN(resid)*g+b [* mask[row]];  optional f16 mirror for next GEMM
__global__ void resid_ln_k(const float* __restrict__ xa, const float* __restrict__ ya,
                           const float* __restrict__ mask, int useMask0,
                           const float* __restrict__ g, const float* __restrict__ bt,
                           float* __restrict__ outp, _Float16* __restrict__ outh,
                           int applyOutMask) {
    int row = blockIdx.x;
    int b = row / NN;
    float f = mask[row] * (useMask0 ? mask[(size_t)b * NN] : 1.0f);
    const float* xr = xa + (size_t)row * DD;
    const float* yr = ya + (size_t)row * DD;
    int t = threadIdx.x;
    float v[4];
    float s = 0.f, s2 = 0.f;
#pragma unroll
    for (int j = 0; j < 4; j++) {
        int c = t + j * 256;
        float u = xr[c] + yr[c] * f;
        v[j] = u; s += u; s2 += u * u;
    }
    s  = blockRedSum(s);
    s2 = blockRedSum(s2);
    float mu  = s * (1.f / DD);
    float var = s2 * (1.f / DD) - mu * mu;
    float rs  = rsqrtf(var + 1e-5f);
    float of  = applyOutMask ? mask[row] : 1.f;
#pragma unroll
    for (int j = 0; j < 4; j++) {
        int c = t + j * 256;
        float o = ((v[j] - mu) * rs * g[c] + bt[c]) * of;
        outp[(size_t)row * DD + c] = o;
        if (outh) outh[(size_t)row * DD + c] = (_Float16)o;
    }
}

// ---------------- host side ----------------
extern "C" void kernel_launch(void* const* d_in, const int* in_sizes, int n_in,
                              void* d_out, int out_size, void* d_ws, size_t ws_size,
                              hipStream_t stream) {
    const float* src   = (const float*)d_in[0];
    const float* mask  = (const float*)d_in[1];
    const float* W_in  = (const float*)d_in[2];
    const float* b_in  = (const float*)d_in[3];
    const float* W_out = (const float*)d_in[4];
    const float* b_out = (const float*)d_in[5];
    const float* W1    = (const float*)d_in[6];
    const float* b1    = (const float*)d_in[7];
    const float* W2    = (const float*)d_in[8];
    const float* b2    = (const float*)d_in[9];
    const float* g1    = (const float*)d_in[10];
    const float* be1   = (const float*)d_in[11];
    const float* g2    = (const float*)d_in[12];
    const float* be2   = (const float*)d_in[13];
    float* out = (float*)d_out;
    float* ws  = (float*)d_ws;
    (void)in_sizes; (void)n_in; (void)out_size; (void)ws_size;

    // ---- workspace layout (float slots), lifetime-based reuse, ~168 MB total
    size_t o = 0;
    float* scal = ws;                          o += 64;
    float* qkvf = ws + o;                      o += (size_t)MROWS * 3 * DD;        // 24 MB f32
    float* awb  = ws + o;                      o += (size_t)BB * HH * NN * NN;     // 64 MB f32
    _Float16* attnq = (_Float16*)(ws + o);     o += (size_t)BB * HH * NN * NN / 2; // 32 MB f16
    _Float16* aof   = (_Float16*)(ws + o);     o += (size_t)MROWS * DD / 2;        // 4 MB f16
    _Float16* xq    = (_Float16*)(ws + o);     o += (size_t)MROWS * DD / 2;        // 4 MB f16
    _Float16* winq  = (_Float16*)(ws + o);     o += (size_t)3 * DD * DD / 2;       // 6 MB f16
    _Float16* qkvq  = (_Float16*)(ws + o);     o += (size_t)MROWS * 3 * DD / 2;    // 12 MB f16 (v third unused)
    _Float16* vt    = (_Float16*)(ws + o);     o += (size_t)BB * HH * DHD * NN / 2;// 4 MB f16 (transposed V)
    _Float16* woutq = (_Float16*)(ws + o);     o += (size_t)DD * DD / 2;           // 2 MB f16
    _Float16* w1q   = (_Float16*)(ws + o);     o += (size_t)FFD * DD / 2;          // 8 MB f16
    _Float16* w2q   = (_Float16*)(ws + o);     o += (size_t)DD * FFD / 2;          // 8 MB f16
    // reuse (sequenced by stream order):
    float*    projb = qkvf;                                 // qkvf dead after vtrans/cvt
    float*    ln1b  = qkvf + (size_t)MROWS * DD;
    _Float16* ln1h  = (_Float16*)(qkvf + (size_t)2 * MROWS * DD);
    _Float16* hb    = (_Float16*)awb;                       // aw dead after PV GEMM
    float*    ffb   = awb + (size_t)MROWS * FFD / 2;        // disjoint from hb

    const long long sQKV = (long long)NN * 3 * DD;          // per-b head-view stride (elements)

    // 0) init scalars; convert unquantized weights to f16 (stream order, independent)
    init_scalars_k<<<1, 32, 0, stream>>>(scal);
    cvt_q_k<<<2048, 256, 0, stream>>>(W_out, DD, DD, DD, woutq, DD, scal, 0, 0);
    cvt_q_k<<<2048, 256, 0, stream>>>(W1, FFD, DD, DD, w1q, DD, scal, 0, 0);
    cvt_q_k<<<2048, 256, 0, stream>>>(W2, DD, FFD, FFD, w2q, FFD, scal, 0, 0);

    // 1) per-tensor ranges + quantized f16 staging of src and W_in
    minmax_k<<<1024, 256, 0, stream>>>(src,  MROWS, DD, DD, scal, 0, 1);
    minmax_k<<<1024, 256, 0, stream>>>(W_in, 3 * DD, DD, DD, scal, 2, 3);
    cvt_q_k<<<2048, 256, 0, stream>>>(src,  MROWS, DD, DD, xq,   DD, scal, 0, 1);
    cvt_q_k<<<2048, 256, 0, stream>>>(W_in, 3 * DD, DD, DD, winq, DD, scal, 2, 1);

    // 2) in_proj: qkv = fq(src) @ fq(W_in)^T + b_in, then *mask[row]
    gemm16_k<<<dim3((3 * DD) / 128, MROWS / 128, 1), 128, 0, stream>>>(
        xq, DD, 0, 0, 1,
        winq, DD, 0, 0, 1,
        qkvf, 3 * DD, 0, 0, 1,
        MROWS, 3 * DD, DD,
        b_in, mask, /*epi=*/1, 1.0f, /*outHalf=*/0);

    // 3) ranges of q, k, v; quantized f16 staging (q,k in-place layout; v transposed per head)
    minmax_k<<<1024, 256, 0, stream>>>(qkvf,          MROWS, DD, 3 * DD, scal, 4, 5);
    minmax_k<<<1024, 256, 0, stream>>>(qkvf + DD,     MROWS, DD, 3 * DD, scal, 6, 7);
    minmax_k<<<1024, 256, 0, stream>>>(qkvf + 2 * DD, MROWS, DD, 3 * DD, scal, 8, 9);
    cvt_q_k<<<2048, 256, 0, stream>>>(qkvf,      MROWS, DD, 3 * DD, qkvq,      3 * DD, scal, 4, 1);
    cvt_q_k<<<2048, 256, 0, stream>>>(qkvf + DD, MROWS, DD, 3 * DD, qkvq + DD, 3 * DD, scal, 6, 1);
    vtrans_k<<<dim3(NN / 16, DHD / 16, BB * HH), dim3(16, 16), 0, stream>>>(qkvf, vt, scal, 8);

    // 4) aw[z] = (fq(q)[z] @ fq(k)[z]^T) * (1/8), z = b*H + h (strided head views)
    gemm16_k<<<dim3(NN / 128, NN / 128, BB * HH), 128, 0, stream>>>(
        qkvq,      3 * DD, sQKV, DHD, HH,
        qkvq + DD, 3 * DD, sQKV, DHD, HH,
        awb, NN, (long long)NN * NN, 0, 1,
        NN, NN, DHD,
        nullptr, nullptr, /*epi=*/0, 0.125f, /*outHalf=*/0);

    // 5) masked softmax in-place + global prob max; quantize probs (always_zero) to f16
    softmax_k<<<BB * HH * NN, 256, 0, stream>>>(awb, mask, scal, 11);
    cvt_q_k<<<4096, 256, 0, stream>>>(awb, BB * HH * NN, NN, NN, attnq, NN, scal, 11, 2);

    // 6) attn_out = fq0(attn) @ fq(v): B = per-head transposed V (contiguous NxK);
    //    merge heads on f16 store
    gemm16_k<<<dim3(1, NN / 128, BB * HH), 128, 0, stream>>>(
        attnq, NN, (long long)NN * NN, 0, 1,
        vt, NN, (long long)DHD * NN, 0, 1,
        (float*)aof, DD, (long long)NN * DD, DHD, HH,
        NN, DHD, NN,
        nullptr, nullptr, /*epi=*/0, 1.0f, /*outHalf=*/1);

    // 7) out_proj: proj = attn_out @ W_out^T + b_out
    gemm16_k<<<dim3(DD / 128, MROWS / 128, 1), 128, 0, stream>>>(
        aof, DD, 0, 0, 1,
        woutq, DD, 0, 0, 1,
        projb, DD, 0, 0, 1,
        MROWS, DD, DD,
        b_out, nullptr, /*epi=*/0, 1.0f, /*outHalf=*/0);

    // 8) resid1 = src + proj * mask[row]*mask[b,0]; ln1 = LN(resid1) (+f16 mirror)
    resid_ln_k<<<MROWS, 256, 0, stream>>>(src, projb, mask, /*useMask0=*/1,
                                          g1, be1, ln1b, ln1h, /*applyOutMask=*/0);

    // 9) FF1: h = relu(ln1 @ W1^T + b1) * mask[row]  (f16 out)
    gemm16_k<<<dim3(FFD / 128, MROWS / 128, 1), 128, 0, stream>>>(
        ln1h, DD, 0, 0, 1,
        w1q, DD, 0, 0, 1,
        (float*)hb, FFD, 0, 0, 1,
        MROWS, FFD, DD,
        b1, mask, /*epi=*/2, 1.0f, /*outHalf=*/1);

    // 10) FF2: ff = h @ W2^T + b2
    gemm16_k<<<dim3(DD / 128, MROWS / 128, 1), 128, 0, stream>>>(
        hb, FFD, 0, 0, 1,
        w2q, FFD, 0, 0, 1,
        ffb, DD, 0, 0, 1,
        MROWS, DD, FFD,
        b2, nullptr, /*epi=*/0, 1.0f, /*outHalf=*/0);

    // 11) out = LN(ln1 + ff*mask[row], g2, be2) * mask[row]
    resid_ln_k<<<MROWS, 256, 0, stream>>>(ln1b, ffb, mask, /*useMask0=*/0,
                                          g2, be2, out, nullptr, /*applyOutMask=*/1);
}
